// LayoutLMv2Encoder_16028817949020
// MI455X (gfx1250) — matrix operand order repaired and checked by reading, not compiled, not run on hardware
//
#include <hip/hip_runtime.h>
#include <math.h>

// ---------------- constants ----------------
#define Bc 8
#define Sq 512
#define Dm 768
#define Hh 12
#define DHh 64
#define Ff 3072
#define Lc 12
#define INV_SQRT_DH 0.125f

typedef __attribute__((ext_vector_type(16))) __bf16 bf16x16;
typedef __attribute__((ext_vector_type(8)))  float  floatx8;
typedef __attribute__((ext_vector_type(4))) unsigned int uint32x4;
typedef __attribute__((ext_vector_type(8))) int int32x8;
typedef __attribute__((ext_vector_type(4))) int int32x4;

union Frag {
    uint4    u2[2];
    __bf16   h[16];
    bf16x16  v;
};

// ---------------- TDM: async 2D bf16 tile load global -> LDS ----------------
// D# per CDNA5 ISA ch.8: group0 {count,lds_addr,global_addr,type=2},
// group1 {data_size=2B, tensor_dim0/1, tile_dim0/1, dim0_stride}, groups2/3 = 0 (2D).
// This toolchain exposes the 6-arg builtin form (clang-23 / therock-10.0 headers).
__device__ __forceinline__ void tdm_load_2d_bf16(unsigned lds_off, const void* gptr,
                                                 unsigned tensor_w, unsigned tensor_h,
                                                 unsigned tile_w, unsigned tile_h,
                                                 unsigned stride0_elems) {
    unsigned long long ga = (unsigned long long)gptr;
    uint32x4 g0;
    g0[0] = 1u;                                             // count=1 (user descriptor)
    g0[1] = lds_off;                                        // LDS byte address
    g0[2] = (unsigned)(ga & 0xffffffffu);                   // global_addr[31:0]
    g0[3] = (unsigned)((ga >> 32) & 0x1ffffffu) | (2u << 30); // addr[56:32] | type=2
    int32x8 g1;
    g1[0] = (int)(1u << 16);                                // wg_mask=0, data_size=1 (2B)
    g1[1] = (int)((tensor_w & 0xffffu) << 16);              // tensor_dim0[15:0]
    g1[2] = (int)(((tensor_w >> 16) & 0xffffu) | ((tensor_h & 0xffffu) << 16));
    g1[3] = (int)(((tensor_h >> 16) & 0xffffu) | ((tile_w & 0xffffu) << 16));
    g1[4] = (int)(tile_h & 0xffffu);                        // tile_dim1, tile_dim2=0
    g1[5] = (int)stride0_elems;                             // tensor_dim0_stride[31:0]
    g1[6] = 0;
    g1[7] = 0;
    int32x4 z4 = (int32x4){0, 0, 0, 0};
    int32x8 z8 = (int32x8){0, 0, 0, 0, 0, 0, 0, 0};
    __builtin_amdgcn_tensor_load_to_lds(g0, g1, z4, z4, z8, 0);
}

__device__ __forceinline__ unsigned lds_offset_of(const void* p) {
    return (unsigned)(unsigned long long)p;   // LDS aperture: addr[31:0] == LDS offset
}

// ---------------- fp32 -> bf16 convert ----------------
__global__ __launch_bounds__(256) void cvt_f32_bf16(const float* __restrict__ in,
                                                    __bf16* __restrict__ out, int n) {
    int i = (blockIdx.x * 256 + threadIdx.x) * 4;
    if (i + 3 < n) {
        float4 f = *(const float4*)(in + i);
        out[i + 0] = (__bf16)f.x;
        out[i + 1] = (__bf16)f.y;
        out[i + 2] = (__bf16)f.z;
        out[i + 3] = (__bf16)f.w;
    }
}

// ---------------- relative-position bias precompute ----------------
__device__ __forceinline__ int rel_bucket(long long rel, int num_buckets, int max_distance) {
    int nb = num_buckets / 2;
    int ret = (rel > 0) ? nb : 0;
    long long n = rel < 0 ? -rel : rel;
    int max_exact = nb / 2;
    int vl;
    if (n < max_exact) {
        vl = (int)n;
    } else {
        long long ns = n < 1 ? 1 : n;
        float t = logf((float)ns / (float)max_exact) /
                  logf((float)max_distance / (float)max_exact) * (float)(nb - max_exact);
        vl = max_exact + (int)t;
        if (vl > nb - 1) vl = nb - 1;
    }
    return ret + vl;
}

__global__ __launch_bounds__(256) void bias_kernel(const long long* __restrict__ pos,
                                                   const long long* __restrict__ bbox,
                                                   const int* __restrict__ mask,
                                                   const float* __restrict__ rp_tab,
                                                   const float* __restrict__ rx_tab,
                                                   const float* __restrict__ ry_tab,
                                                   float* __restrict__ bias) {
    int t = threadIdx.x;
    int i = blockIdx.y * 16 + (t >> 4);
    int j = blockIdx.x * 16 + (t & 15);
    int b = blockIdx.z;

    long long pi = pos[b * Sq + i], pj = pos[b * Sq + j];
    int rp = rel_bucket(pj - pi, 32, 128);

    long long cxi = bbox[((size_t)(b * Sq + i)) * 4 + 0];
    long long cxj = bbox[((size_t)(b * Sq + j)) * 4 + 0];
    long long cyi = bbox[((size_t)(b * Sq + i)) * 4 + 3];
    long long cyj = bbox[((size_t)(b * Sq + j)) * 4 + 3];
    int bx = rel_bucket(cxj - cxi, 64, 256);
    int by = rel_bucket(cyj - cyi, 64, 256);

    float m = (mask[b * Sq + j] != 0) ? -1e30f : 0.0f;
    for (int h = 0; h < Hh; ++h) {
        float v = rp_tab[rp * Hh + h] + rx_tab[bx * Hh + h] + ry_tab[by * Hh + h] + m;
        bias[(((size_t)(b * Hh + h)) * Sq + i) * Sq + j] = v;
    }
}

// ---------------- tiled WMMA GEMM: C = A(MxK,bf16) * B(KxN,bf16) ----------------
// Block tile 128x128, 8 waves (4 in M x 2 in N), wave tile 32x64, K-step 32.
// A tile double-buffered via TDM (tensor_load_to_lds); B tile staged transposed by VALU.
enum { EPI_F32 = 0, EPI_QKV = 1, EPI_GELU = 2 };

template <int EPI>
__global__ __launch_bounds__(256) void gemm_bf16(const __bf16* __restrict__ A,
                                                 const __bf16* __restrict__ B,
                                                 int M, int N, int K,
                                                 const float* __restrict__ bias1,
                                                 const float* __restrict__ bias2,
                                                 float scale,
                                                 float* __restrict__ Cf,
                                                 __bf16* __restrict__ Cb) {
    __shared__ __bf16 As[2][128][32];   // [buf][m][k]  (TDM destination)
    __shared__ __bf16 Bs[2][128][32];   // [buf][n][k]  (transposed)

    const int t    = threadIdx.x;
    const int lane = t & 31;
    const int wave = t >> 5;
    const int lm   = lane & 15;
    const int half = lane >> 4;
    const int wm   = wave & 3;       // 0..3 -> 32-row slice
    const int wn   = wave >> 2;      // 0..1 -> 64-col slice

    const int m0 = blockIdx.y * 128;
    const int n0 = blockIdx.x * 128;

    const int b_row = t >> 3, b_col = (t & 7) * 16;

    floatx8 acc[2][4];
    for (int mi = 0; mi < 2; ++mi)
        for (int ni = 0; ni < 4; ++ni)
            acc[mi][ni] = (floatx8){0, 0, 0, 0, 0, 0, 0, 0};

    // ---- prologue: stage k0 = 0 into buffer 0 ----
    if (wave == 0) {
        tdm_load_2d_bf16(lds_offset_of(&As[0][0][0]),
                         A + (size_t)m0 * K, (unsigned)K, (unsigned)(M - m0),
                         32u, 128u, (unsigned)K);
    }
    {
        const __bf16* src = B + (size_t)b_row * N + n0 + b_col;
        Frag f;
        f.u2[0] = *(const uint4*)(src + 0);
        f.u2[1] = *(const uint4*)(src + 8);
        #pragma unroll
        for (int e = 0; e < 16; ++e) Bs[0][b_col + e][b_row] = f.h[e];
    }

    for (int k0 = 0, it = 0; k0 < K; k0 += 32, ++it) {
        const int cur = it & 1, nxt = cur ^ 1;
        const bool has_next = (k0 + 32 < K);
        __syncthreads();   // previous iteration's reads of buffers [nxt] complete

        if (has_next) {
            if (wave == 0) {
                tdm_load_2d_bf16(lds_offset_of(&As[nxt][0][0]),
                                 A + (size_t)m0 * K + k0 + 32, (unsigned)K,
                                 (unsigned)(M - m0), 32u, 128u, (unsigned)K);
            }
            const __bf16* src = B + (size_t)(k0 + 32 + b_row) * N + n0 + b_col;
            Frag f;
            f.u2[0] = *(const uint4*)(src + 0);
            f.u2[1] = *(const uint4*)(src + 8);
            #pragma unroll
            for (int e = 0; e < 16; ++e) Bs[nxt][b_col + e][b_row] = f.h[e];
            __builtin_prefetch(B + (size_t)(k0 + 64 + b_row) * N + n0 + b_col, 0, 0);
        }
        // TDM in-order: after issuing tile it+1, "<=1 outstanding" => tile it landed
        if (wave == 0) {
            if (has_next) __builtin_amdgcn_s_wait_tensorcnt(1);
            else          __builtin_amdgcn_s_wait_tensorcnt(0);
        }
        __syncthreads();

        #pragma unroll
        for (int mi = 0; mi < 2; ++mi) {
            const int m = wm * 32 + mi * 16 + lm;
            Frag a;
            a.u2[0] = *(const uint4*)&As[cur][m][half * 8];
            a.u2[1] = *(const uint4*)&As[cur][m][16 + half * 8];
            #pragma unroll
            for (int ni = 0; ni < 4; ++ni) {
                const int n = wn * 64 + ni * 16 + lm;
                Frag bf;
                bf.u2[0] = *(const uint4*)&Bs[cur][n][half * 8];
                bf.u2[1] = *(const uint4*)&Bs[cur][n][16 + half * 8];
                acc[mi][ni] = __builtin_amdgcn_wmma_f32_16x16x32_bf16(
                    false, a.v, false, bf.v, (short)0, acc[mi][ni], false, false);
            }
        }
    }

    // epilogue
    #pragma unroll
    for (int mi = 0; mi < 2; ++mi) {
        #pragma unroll
        for (int ni = 0; ni < 4; ++ni) {
            const int n = n0 + wn * 64 + ni * 16 + lm;
            #pragma unroll
            for (int r = 0; r < 8; ++r) {
                const int m = m0 + wm * 32 + mi * 16 + r + 8 * half;
                float v = acc[mi][ni][r];
                if (EPI == EPI_F32) {
                    Cf[(size_t)m * N + n] = v;
                } else if (EPI == EPI_QKV) {
                    if (n < Dm)             v = (v + bias1[n]) * scale;        // q
                    else if (n >= 2 * Dm)   v = v + bias2[n - 2 * Dm];          // v
                    Cb[(size_t)m * N + n] = (__bf16)v;
                } else { // EPI_GELU
                    v += bias1[n];
                    v = 0.5f * v * (1.0f + erff(v * 0.70710678118f));
                    Cb[(size_t)m * N + n] = (__bf16)v;
                }
            }
        }
    }
}

// ---------------- fused attention: one block per (b, h, 16 query rows) ----------------
__global__ __launch_bounds__(256) void attn_kernel(const __bf16* __restrict__ qkv,
                                                   const float* __restrict__ bias,
                                                   __bf16* __restrict__ ctx) {
    __shared__ struct {
        float  S[16][512];    // 32 KB scores
        __bf16 P[16][512];    // 16 KB probs
        __bf16 Vt[128][64];   // 16 KB V chunk (row-major [j][d], TDM destination)
    } sh;                      // 64 KB total

    const int i0 = blockIdx.x * 16;
    const int h  = blockIdx.y;
    const int b  = blockIdx.z;
    const int t    = threadIdx.x;
    const int lane = t & 31;
    const int wave = t >> 5;
    const int lm   = lane & 15;
    const int half = lane >> 4;

    // ---- phase 1: scores = Q K^T (scale folded into q) + bias ----
    const int j0 = wave * 64;
    floatx8 sacc[4];
    for (int nt = 0; nt < 4; ++nt) sacc[nt] = (floatx8){0, 0, 0, 0, 0, 0, 0, 0};

    const __bf16* qrow = qkv + (size_t)(b * Sq + i0 + lm) * (3 * Dm) + h * DHh;
    #pragma unroll
    for (int d0 = 0; d0 < DHh; d0 += 32) {
        Frag a;
        a.u2[0] = *(const uint4*)(qrow + d0 + half * 8);
        a.u2[1] = *(const uint4*)(qrow + d0 + 16 + half * 8);
        #pragma unroll
        for (int nt = 0; nt < 4; ++nt) {
            const __bf16* krow = qkv + (size_t)(b * Sq + j0 + nt * 16 + lm) * (3 * Dm)
                                 + Dm + h * DHh + d0 + half * 8;
            Frag bf;
            bf.u2[0] = *(const uint4*)(krow + 0);
            bf.u2[1] = *(const uint4*)(krow + 16);
            sacc[nt] = __builtin_amdgcn_wmma_f32_16x16x32_bf16(
                false, a.v, false, bf.v, (short)0, sacc[nt], false, false);
        }
    }
    #pragma unroll
    for (int nt = 0; nt < 4; ++nt) {
        const int jn = j0 + nt * 16 + lm;
        #pragma unroll
        for (int r = 0; r < 8; ++r) {
            const int m = r + 8 * half;
            sh.S[m][jn] = sacc[nt][r] +
                bias[(((size_t)(b * Hh + h)) * Sq + i0 + m) * Sq + jn];
        }
    }
    __syncthreads();

    // ---- softmax over j (rows of 512), 16 threads per row ----
    float* red = (float*)&sh.Vt[0][0];  // reuse Vt space for reductions
    const int row = t >> 4, sub = t & 15;
    float lmax = -3.4e38f;
    for (int j = sub * 32; j < sub * 32 + 32; ++j) lmax = fmaxf(lmax, sh.S[row][j]);
    red[t] = lmax;
    __syncthreads();
    if (sub == 0) {
        float mx = red[row * 16];
        for (int q = 1; q < 16; ++q) mx = fmaxf(mx, red[row * 16 + q]);
        red[256 + row] = mx;
    }
    __syncthreads();
    const float mx = red[256 + row];
    float lsum = 0.0f;
    for (int j = sub * 32; j < sub * 32 + 32; ++j) {
        float e = __expf(sh.S[row][j] - mx);
        sh.S[row][j] = e;
        lsum += e;
    }
    red[t] = lsum;
    __syncthreads();
    if (sub == 0) {
        float s = red[row * 16];
        for (int q = 1; q < 16; ++q) s += red[row * 16 + q];
        red[288 + row] = s;
    }
    __syncthreads();
    const float inv = 1.0f / red[288 + row];
    for (int j = sub * 32; j < sub * 32 + 32; ++j)
        sh.P[row][j] = (__bf16)(sh.S[row][j] * inv);
    __syncthreads();   // reductions done; Vt free for TDM from here

    // ---- phase 2: ctx = P @ V; V staged per 128-row chunk via TDM ----
    floatx8 vacc = (floatx8){0, 0, 0, 0, 0, 0, 0, 0};
    for (int c = 0; c < 4; ++c) {
        if (wave == 0) {
            // 128 x 64 bf16 tile of V (row stride 3*Dm elements) -> sh.Vt
            tdm_load_2d_bf16(lds_offset_of(&sh.Vt[0][0]),
                             qkv + (size_t)(b * Sq + c * 128) * (3 * Dm) + 2 * Dm + h * DHh,
                             64u, 128u, 64u, 128u, (unsigned)(3 * Dm));
            __builtin_amdgcn_s_wait_tensorcnt(0);
        }
        __syncthreads();
        if (wave < 4) {
            const int d0 = wave * 16;
            #pragma unroll
            for (int ks = 0; ks < 4; ++ks) {
                const int kb = c * 128 + ks * 32;
                Frag a;
                a.u2[0] = *(const uint4*)&sh.P[lm][kb + half * 8];
                a.u2[1] = *(const uint4*)&sh.P[lm][kb + 16 + half * 8];
                Frag bf;
                #pragma unroll
                for (int e = 0; e < 16; e += 2) {
                    const int k = ks * 32 + (e < 8 ? half * 8 + e : 16 + half * 8 + (e - 8));
                    bf.h[e]     = sh.Vt[k][d0 + lm];
                    bf.h[e + 1] = sh.Vt[k + 1][d0 + lm];
                }
                vacc = __builtin_amdgcn_wmma_f32_16x16x32_bf16(
                    false, a.v, false, bf.v, (short)0, vacc, false, false);
            }
        }
        __syncthreads();
    }
    if (wave < 4) {
        const int d0 = wave * 16;
        #pragma unroll
        for (int r = 0; r < 8; ++r) {
            const int m = r + 8 * half;
            ctx[(size_t)(b * Sq + i0 + m) * Dm + h * DHh + d0 + lm] = (__bf16)vacc[r];
        }
    }
}

// ---------------- fused bias + residual + LayerNorm (emits fp32 + bf16) ----------------
__global__ __launch_bounds__(256) void ln_kernel(const float* __restrict__ x,
                                                 const float* __restrict__ bias,
                                                 const float* __restrict__ res,
                                                 const float* __restrict__ g,
                                                 const float* __restrict__ bv,
                                                 float* __restrict__ outf,
                                                 __bf16* __restrict__ outb) {
    __shared__ float rs[256], rs2[256];
    const size_t row = blockIdx.x;
    const int t = threadIdx.x;
    float v[3];
    float s = 0.0f, s2 = 0.0f;
    #pragma unroll
    for (int q = 0; q < 3; ++q) {
        const int d = t + q * 256;
        float f = x[row * Dm + d] + bias[d] + res[row * Dm + d];
        v[q] = f;
        s += f;
        s2 += f * f;
    }
    rs[t] = s; rs2[t] = s2;
    __syncthreads();
    for (int off = 128; off > 0; off >>= 1) {
        if (t < off) { rs[t] += rs[t + off]; rs2[t] += rs2[t + off]; }
        __syncthreads();
    }
    const float mean = rs[0] / (float)Dm;
    const float var  = rs2[0] / (float)Dm - mean * mean;
    const float rstd = rsqrtf(var + 1e-12f);
    #pragma unroll
    for (int q = 0; q < 3; ++q) {
        const int d = t + q * 256;
        float o = (v[q] - mean) * rstd * g[d] + bv[d];
        outf[row * Dm + d] = o;
        outb[row * Dm + d] = (__bf16)o;
    }
}

// ---------------- host launch ----------------
extern "C" void kernel_launch(void* const* d_in, const int* in_sizes, int n_in,
                              void* d_out, int out_size, void* d_ws, size_t ws_size,
                              hipStream_t stream) {
    const float*     hs     = (const float*)d_in[0];
    const int*       amask  = (const int*)d_in[1];
    const long long* pos    = (const long long*)d_in[2];
    const long long* bbox   = (const long long*)d_in[3];
    const float*     qkv_w  = (const float*)d_in[4];
    const float*     q_bias = (const float*)d_in[5];
    const float*     v_bias = (const float*)d_in[6];
    const float*     ao_w   = (const float*)d_in[7];
    const float*     ao_b   = (const float*)d_in[8];
    const float*     ln1g   = (const float*)d_in[9];
    const float*     ln1b   = (const float*)d_in[10];
    const float*     in_w   = (const float*)d_in[11];
    const float*     in_b   = (const float*)d_in[12];
    const float*     fo_w   = (const float*)d_in[13];
    const float*     fo_b   = (const float*)d_in[14];
    const float*     ln2g   = (const float*)d_in[15];
    const float*     ln2b   = (const float*)d_in[16];
    const float*     rp_tab = (const float*)d_in[17];
    const float*     rx_tab = (const float*)d_in[18];
    const float*     ry_tab = (const float*)d_in[19];

    const size_t M = (size_t)Bc * Sq;                 // 4096 rows
    size_t off = 0;
    auto carve = [&](size_t bytes) -> char* {
        char* p = (char*)d_ws + off;
        off += (bytes + 255) & ~(size_t)255;
        return p;
    };
    __bf16* wq  = (__bf16*)carve((size_t)Lc * Dm * 3 * Dm * 2);
    __bf16* wa  = (__bf16*)carve((size_t)Lc * Dm * Dm * 2);
    __bf16* wi  = (__bf16*)carve((size_t)Lc * Dm * Ff * 2);
    __bf16* wf  = (__bf16*)carve((size_t)Lc * Ff * Dm * 2);
    float*  bbuf   = (float*)carve((size_t)Bc * Hh * Sq * Sq * 4);
    __bf16* qkvb   = (__bf16*)carve(M * 3 * Dm * 2);
    __bf16* ctxb   = (__bf16*)carve(M * Dm * 2);
    __bf16* hidb   = (__bf16*)carve(M * Dm * 2);
    __bf16* attnb  = (__bf16*)carve(M * Dm * 2);
    __bf16* ffb    = (__bf16*)carve(M * Ff * 2);
    float*  tmpf   = (float*)carve(M * Dm * 4);
    float*  attnf  = (float*)carve(M * Dm * 4);
    float*  hidA   = (float*)carve(M * Dm * 4);
    float*  hidB   = (float*)carve(M * Dm * 4);

    // ---- one-time conversions ----
    auto cvt = [&](const float* src, __bf16* dst, size_t n) {
        cvt_f32_bf16<<<(unsigned)((n / 4 + 255) / 256), 256, 0, stream>>>(src, dst, (int)n);
    };
    cvt(qkv_w, wq, (size_t)Lc * Dm * 3 * Dm);
    cvt(ao_w,  wa, (size_t)Lc * Dm * Dm);
    cvt(in_w,  wi, (size_t)Lc * Dm * Ff);
    cvt(fo_w,  wf, (size_t)Lc * Ff * Dm);
    cvt(hs, hidb, M * Dm);

    bias_kernel<<<dim3(Sq / 16, Sq / 16, Bc), 256, 0, stream>>>(
        pos, bbox, amask, rp_tab, rx_tab, ry_tab, bbuf);

    const float* resf = hs;   // residual source for layer 0
    for (int l = 0; l < Lc; ++l) {
        const __bf16* lwq = wq + (size_t)l * Dm * 3 * Dm;
        const __bf16* lwa = wa + (size_t)l * Dm * Dm;
        const __bf16* lwi = wi + (size_t)l * Dm * Ff;
        const __bf16* lwf = wf + (size_t)l * Ff * Dm;

        // QKV projection (bias + q-scale fused, bf16 out)
        gemm_bf16<EPI_QKV><<<dim3(3 * Dm / 128, (unsigned)(M / 128)), 256, 0, stream>>>(
            hidb, lwq, (int)M, 3 * Dm, Dm,
            q_bias + (size_t)l * Dm, v_bias + (size_t)l * Dm, INV_SQRT_DH,
            nullptr, qkvb);

        // fused attention
        attn_kernel<<<dim3(Sq / 16, Hh, Bc), 256, 0, stream>>>(qkvb, bbuf, ctxb);

        // attention output projection (fp32 out, bias folded into LN)
        gemm_bf16<EPI_F32><<<dim3(Dm / 128, (unsigned)(M / 128)), 256, 0, stream>>>(
            ctxb, lwa, (int)M, Dm, Dm, nullptr, nullptr, 1.0f, tmpf, nullptr);

        ln_kernel<<<(unsigned)M, 256, 0, stream>>>(
            tmpf, ao_b + (size_t)l * Dm, resf,
            ln1g + (size_t)l * Dm, ln1b + (size_t)l * Dm, attnf, attnb);

        // intermediate (bias + exact GELU fused, bf16 out)
        gemm_bf16<EPI_GELU><<<dim3(Ff / 128, (unsigned)(M / 128)), 256, 0, stream>>>(
            attnb, lwi, (int)M, Ff, Dm, in_b + (size_t)l * Ff, nullptr, 1.0f,
            nullptr, ffb);

        // FFN output projection (fp32 out)
        gemm_bf16<EPI_F32><<<dim3(Dm / 128, (unsigned)(M / 128)), 256, 0, stream>>>(
            ffb, lwf, (int)M, Dm, Ff, nullptr, nullptr, 1.0f, tmpf, nullptr);

        float* outf = (l == Lc - 1) ? (float*)d_out : ((l & 1) ? hidB : hidA);
        ln_kernel<<<(unsigned)M, 256, 0, stream>>>(
            tmpf, fo_b + (size_t)l * Dm, attnf,
            ln2g + (size_t)l * Dm, ln2b + (size_t)l * Dm, outf, hidb);
        resf = outf;
    }
}
